// BinaryAttention_50285477101621
// MI455X (gfx1250) — matrix operand order repaired and checked by reading, not compiled, not run on hardware
//
#include <hip/hip_runtime.h>
#include <hip/hip_bf16.h>

// ---------------------------------------------------------------------------
// Types
// ---------------------------------------------------------------------------
typedef __bf16 bf16_t;
typedef __attribute__((ext_vector_type(4)))  __bf16 bf16x4;
typedef __attribute__((ext_vector_type(8)))  __bf16 bf16x8;
typedef __attribute__((ext_vector_type(16))) __bf16 bf16x16;
typedef __attribute__((ext_vector_type(8)))  float  f32x8;

union ABfrag {
    bf16x16 f;
    bf16x8  h[2];
};

__device__ __forceinline__ bf16_t f2bf(float f) {
    unsigned u = __builtin_bit_cast(unsigned, f);
    unsigned r = u + 0x7FFFu + ((u >> 16) & 1u);   // round-to-nearest-even
    unsigned short hv = (unsigned short)(r >> 16);
    return __builtin_bit_cast(bf16_t, hv);
}

// CDNA5 async memory->LDS copy (ASYNCcnt-tracked), inline asm since the
// builtins differ across toolchains.  lds_ptr low 32 bits are the LDS offset.
__device__ __forceinline__ void async_ld_b128(void* lds_ptr, const void* gptr) {
    unsigned l = (unsigned)(uintptr_t)lds_ptr;
    asm volatile("global_load_async_to_lds_b128 %0, %1, off"
                 :: "v"(l), "v"(gptr) : "memory");
}
__device__ __forceinline__ void wait_async0() {
    asm volatile("s_wait_asynccnt 0x0" ::: "memory");
}

#define DIM   1024
#define BATCH 4
#define SEQ   2048
#define HEADS 16
#define DH    64
#define MTOT  (BATCH * SEQ)   // 8192

// ---------------------------------------------------------------------------
// Kernel 1: binarize weights:  Wb[o][c] = mean(|W[o,:]|) * sign(W[o][c])  (bf16)
// one wave per output row
// ---------------------------------------------------------------------------
__global__ void __launch_bounds__(256)
binarize_kernel(const float* __restrict__ W0, const float* __restrict__ W1,
                const float* __restrict__ W2, const float* __restrict__ W3,
                bf16_t* __restrict__ B0, bf16_t* __restrict__ B1,
                bf16_t* __restrict__ B2, bf16_t* __restrict__ B3) {
    const float* W; bf16_t* Bo;
    switch (blockIdx.y) {
        case 0:  W = W0; Bo = B0; break;
        case 1:  W = W1; Bo = B1; break;
        case 2:  W = W2; Bo = B2; break;
        default: W = W3; Bo = B3; break;
    }
    const int wave = threadIdx.x >> 5;
    const int lane = threadIdx.x & 31;
    const int row  = blockIdx.x * 8 + wave;
    const float* wr = W + (size_t)row * DIM;

    float s = 0.0f;
    #pragma unroll
    for (int i = 0; i < DIM / 32; ++i) s += fabsf(wr[lane + 32 * i]);
    #pragma unroll
    for (int m = 16; m > 0; m >>= 1) s += __shfl_xor(s, m, 32);
    const float alpha = s * (1.0f / (float)DIM);

    bf16_t* br = Bo + (size_t)row * DIM;
    #pragma unroll
    for (int i = 0; i < DIM / 32; ++i) {
        const int c = lane + 32 * i;
        const float w = wr[c];
        const float v = (w > 0.0f) ? alpha : ((w < 0.0f) ? -alpha : 0.0f);
        br[c] = f2bf(v);
    }
}

// ---------------------------------------------------------------------------
// Kernel 2: f32 -> bf16 activation convert
// ---------------------------------------------------------------------------
__global__ void __launch_bounds__(256)
cvt_bf16_kernel(const float* __restrict__ x, bf16_t* __restrict__ xb, int n) {
    const int i = (blockIdx.x * blockDim.x + threadIdx.x) * 4;
    if (i < n) {
        const float4 f = *(const float4*)(x + i);
        bf16x4 o;
        o[0] = f2bf(f.x); o[1] = f2bf(f.y); o[2] = f2bf(f.z); o[3] = f2bf(f.w);
        *(bf16x4*)(xb + i) = o;
    }
}

// ---------------------------------------------------------------------------
// Kernel 3: WMMA GEMM:  out[m,n] = sum_c A[m,c] * Bw[n,c] + bias[n]
//   A:  [M x 1024] bf16 row-major (contraction contiguous)
//   Bw: [1024 x 1024] bf16 row-major (contraction contiguous -> perfect B frag)
// Block tile 128(M) x 128(N), 8 waves (4x2), each wave 32x64 (2x4 wmma tiles).
// Tiles staged with double-buffered GLOBAL_LOAD_ASYNC_TO_LDS_B128 (ASYNCcnt).
// TRANS_OUT: store D^T as Vt[b][n][t] (contiguous 8-elem lanes stores).
// ---------------------------------------------------------------------------
template <bool TRANS_OUT, bool OUT_F32>
__global__ void __launch_bounds__(256)
gemm_bf16_kernel(const bf16_t* __restrict__ A, const bf16_t* __restrict__ Bw,
                 const float* __restrict__ bias, void* __restrict__ outv, int M) {
    __shared__ bf16_t As[2][128 * 32];   // 2 x 8 KB
    __shared__ bf16_t Bs[2][128 * 32];   // 2 x 8 KB

    const int t     = threadIdx.x;
    const int lane  = t & 31;
    const int w     = t >> 5;
    const int wy    = w >> 1;         // 0..3 -> M sub-tile (32 rows)
    const int wx    = w & 1;          // 0..1 -> N sub-tile (64 cols)
    const int mbase = blockIdx.x * 128;
    const int nbase = blockIdx.y * 128;
    const int r     = lane & 15;
    const int sel   = lane >> 4;

    f32x8 acc[2][4];
    #pragma unroll
    for (int mt = 0; mt < 2; ++mt)
        #pragma unroll
        for (int nt = 0; nt < 4; ++nt)
            #pragma unroll
            for (int e = 0; e < 8; ++e) acc[mt][nt][e] = 0.0f;

    // async stage one 128x32 A tile + 128x32 B tile into buffer `buf`
    auto issue_tiles = [&](int buf, int k0) {
        #pragma unroll
        for (int j = 0; j < 2; ++j) {
            const int sl  = t + 256 * j;     // 512 slots x 16B = 8 KB per tile
            const int row = sl >> 2;
            const int q   = sl & 3;
            async_ld_b128(&As[buf][row * 32 + q * 8],
                          &A[(size_t)(mbase + row) * DIM + k0 + q * 8]);
            async_ld_b128(&Bs[buf][row * 32 + q * 8],
                          &Bw[(size_t)(nbase + row) * DIM + k0 + q * 8]);
        }
    };

    issue_tiles(0, 0);

    for (int i = 0; i < DIM / 32; ++i) {
        const int cur = i & 1;
        wait_async0();        // my async loads (current buffer) landed in LDS
        __syncthreads();      // everyone's landed; previous compute done
        if (i + 1 < DIM / 32) issue_tiles(cur ^ 1, (i + 1) * 32);

        // fragments from LDS
        ABfrag af[2];
        bf16x16 bfg[4];
        #pragma unroll
        for (int mt = 0; mt < 2; ++mt) {
            const int row = wy * 32 + mt * 16 + r;
            af[mt].h[0] = *(const bf16x8*)&As[cur][row * 32 + sel * 16];
            af[mt].h[1] = *(const bf16x8*)&As[cur][row * 32 + sel * 16 + 8];
        }
        #pragma unroll
        for (int nt = 0; nt < 4; ++nt) {
            const int row = wx * 64 + nt * 16 + r;
            bfg[nt] = *(const bf16x16*)&Bs[cur][row * 32 + sel * 16];
        }
        #pragma unroll
        for (int mt = 0; mt < 2; ++mt)
            #pragma unroll
            for (int nt = 0; nt < 4; ++nt)
                acc[mt][nt] = __builtin_amdgcn_wmma_f32_16x16x32_bf16(
                    false, af[mt].f, false, bfg[nt], (short)0, acc[mt][nt],
                    false, false);
    }

    // epilogue: bias + store
    #pragma unroll
    for (int mt = 0; mt < 2; ++mt) {
        #pragma unroll
        for (int nt = 0; nt < 4; ++nt) {
            const int col  = nbase + wx * 64 + nt * 16 + r;
            const float bc = bias[col];
            #pragma unroll
            for (int e = 0; e < 8; ++e) {
                const int row   = mbase + wy * 32 + mt * 16 + sel * 8 + e;
                const float val = acc[mt][nt][e] + bc;
                if (TRANS_OUT) {
                    const int bb = row >> 11;      // /SEQ
                    const int tt = row & (SEQ - 1);
                    ((bf16_t*)outv)[((size_t)(bb * DIM + col)) * SEQ + tt] = f2bf(val);
                } else if (OUT_F32) {
                    ((float*)outv)[(size_t)row * DIM + col] = val;
                } else {
                    ((bf16_t*)outv)[(size_t)row * DIM + col] = f2bf(val);
                }
            }
        }
    }
}

// ---------------------------------------------------------------------------
// Kernel 4: flash attention. One wave handles 16 queries of one (b,h).
//   Q,K: [B*T, 1024] bf16 row-major.  Vt: [B, 1024, T] bf16 (channel-major).
//   Key tiles of 32; online softmax; P transposed through LDS (C->A layout).
// ---------------------------------------------------------------------------
__global__ void __launch_bounds__(32)
flash_attn_kernel(const bf16_t* __restrict__ Q, const bf16_t* __restrict__ K,
                  const bf16_t* __restrict__ Vt, bf16_t* __restrict__ Y) {
    __shared__ bf16_t Pb[16 * 32];    // P staging: 16 q-rows x 32 keys

    const int lane = threadIdx.x;
    const int r    = lane & 15;
    const int sel  = lane >> 4;
    const int b    = blockIdx.z;
    const int h    = blockIdx.y;
    const int q0   = blockIdx.x * 16;
    const float scale = 0.125f;       // Dh^-0.5 = 64^-0.5

    // Q A-fragments (Dh=64 -> two k=32 fragments), straight from global
    const bf16_t* qrow = Q + ((size_t)(b * SEQ + q0 + r)) * DIM + h * DH;
    ABfrag qa[2];
    #pragma unroll
    for (int c = 0; c < 2; ++c) {
        qa[c].h[0] = *(const bf16x8*)(qrow + c * 32 + sel * 16);
        qa[c].h[1] = *(const bf16x8*)(qrow + c * 32 + sel * 16 + 8);
    }

    f32x8 o[4], mrun, lrun;
    #pragma unroll
    for (int d = 0; d < 4; ++d)
        #pragma unroll
        for (int e = 0; e < 8; ++e) o[d][e] = 0.0f;
    #pragma unroll
    for (int e = 0; e < 8; ++e) { mrun[e] = -1e30f; lrun[e] = 0.0f; }

    for (int kt = 0; kt < SEQ; kt += 32) {
        // ---- S = scale * Q K^T  (two 16x16 col tiles) ----
        f32x8 s[2];
        #pragma unroll
        for (int j = 0; j < 2; ++j) {
            #pragma unroll
            for (int e = 0; e < 8; ++e) s[j][e] = 0.0f;
            const int key = kt + j * 16 + r;
            #pragma unroll
            for (int c = 0; c < 2; ++c) {
                const bf16x16 kf = *(const bf16x16*)
                    &K[((size_t)(b * SEQ + key)) * DIM + h * DH + c * 32 + sel * 16];
                s[j] = __builtin_amdgcn_wmma_f32_16x16x32_bf16(
                    false, qa[c].f, false, kf, (short)0, s[j], false, false);
            }
            #pragma unroll
            for (int e = 0; e < 8; ++e) s[j][e] *= scale;
        }

        // ---- online softmax (rows striped: lanes 0-15 rows 0-7, 16-31 rows 8-15)
        f32x8 mx;
        #pragma unroll
        for (int e = 0; e < 8; ++e) mx[e] = fmaxf(s[0][e], s[1][e]);
        #pragma unroll
        for (int msk = 1; msk < 16; msk <<= 1)
            #pragma unroll
            for (int e = 0; e < 8; ++e) mx[e] = fmaxf(mx[e], __shfl_xor(mx[e], msk, 32));

        f32x8 corr, rs;
        #pragma unroll
        for (int e = 0; e < 8; ++e) {
            const float mn = fmaxf(mrun[e], mx[e]);
            corr[e] = __expf(mrun[e] - mn);
            mrun[e] = mn;
        }
        #pragma unroll
        for (int j = 0; j < 2; ++j)
            #pragma unroll
            for (int e = 0; e < 8; ++e) s[j][e] = __expf(s[j][e] - mrun[e]);
        #pragma unroll
        for (int e = 0; e < 8; ++e) rs[e] = s[0][e] + s[1][e];
        #pragma unroll
        for (int msk = 1; msk < 16; msk <<= 1)
            #pragma unroll
            for (int e = 0; e < 8; ++e) rs[e] += __shfl_xor(rs[e], msk, 32);
        #pragma unroll
        for (int e = 0; e < 8; ++e) lrun[e] = lrun[e] * corr[e] + rs[e];
        #pragma unroll
        for (int d = 0; d < 4; ++d)
            #pragma unroll
            for (int e = 0; e < 8; ++e) o[d][e] *= corr[e];

        // ---- P: C-layout -> A-layout through LDS ----
        #pragma unroll
        for (int j = 0; j < 2; ++j)
            #pragma unroll
            for (int e = 0; e < 8; ++e)
                Pb[(sel * 8 + e) * 32 + j * 16 + r] = f2bf(s[j][e]);
        __syncthreads();   // single-wave WG: cheap ordering point
        ABfrag pf;
        pf.h[0] = *(const bf16x8*)&Pb[r * 32 + sel * 16];
        pf.h[1] = *(const bf16x8*)&Pb[r * 32 + sel * 16 + 8];
        __syncthreads();

        // ---- O += P V  (Vt channel-major -> contiguous B fragments) ----
        #pragma unroll
        for (int d = 0; d < 4; ++d) {
            const bf16x16 vf = *(const bf16x16*)
                &Vt[((size_t)(b * DIM + h * DH + d * 16 + r)) * SEQ + kt + sel * 16];
            o[d] = __builtin_amdgcn_wmma_f32_16x16x32_bf16(
                false, pf.f, false, vf, (short)0, o[d], false, false);
        }
    }

    // ---- normalize and store Y (bf16, [B*T, 1024]) ----
    #pragma unroll
    for (int d = 0; d < 4; ++d) {
        const int col = h * DH + d * 16 + r;
        #pragma unroll
        for (int e = 0; e < 8; ++e) {
            const int row = q0 + sel * 8 + e;
            Y[((size_t)(b * SEQ + row)) * DIM + col] = f2bf(o[d][e] / lrun[e]);
        }
    }
}

// ---------------------------------------------------------------------------
// Launch
// ---------------------------------------------------------------------------
extern "C" void kernel_launch(void* const* d_in, const int* in_sizes, int n_in,
                              void* d_out, int out_size, void* d_ws, size_t ws_size,
                              hipStream_t stream) {
    (void)in_sizes; (void)n_in; (void)out_size; (void)ws_size;
    const float* x  = (const float*)d_in[0];
    const float* Wq = (const float*)d_in[1];
    const float* bq = (const float*)d_in[2];
    const float* Wk = (const float*)d_in[3];
    const float* bk = (const float*)d_in[4];
    const float* Wv = (const float*)d_in[5];
    const float* bv = (const float*)d_in[6];
    const float* Wp = (const float*)d_in[7];
    const float* bp = (const float*)d_in[8];
    float* out = (float*)d_out;

    char* ws = (char*)d_ws;
    const size_t MB = 1024 * 1024;
    bf16_t* xb  = (bf16_t*)(ws +  0 * MB);  // 16 MB  x in bf16
    bf16_t* Wqb = (bf16_t*)(ws + 16 * MB);  //  2 MB
    bf16_t* Wkb = (bf16_t*)(ws + 18 * MB);  //  2 MB
    bf16_t* Wvb = (bf16_t*)(ws + 20 * MB);  //  2 MB
    bf16_t* Wpb = (bf16_t*)(ws + 22 * MB);  //  2 MB
    bf16_t* Qb  = (bf16_t*)(ws + 24 * MB);  // 16 MB  [B*T, C]
    bf16_t* Kb  = (bf16_t*)(ws + 40 * MB);  // 16 MB  [B*T, C]
    bf16_t* Vt  = (bf16_t*)(ws + 56 * MB);  // 16 MB  [B, C, T]
    bf16_t* Yb  = (bf16_t*)(ws + 72 * MB);  // 16 MB  [B*T, C]

    binarize_kernel<<<dim3(128, 4), 256, 0, stream>>>(Wq, Wk, Wv, Wp,
                                                      Wqb, Wkb, Wvb, Wpb);
    cvt_bf16_kernel<<<(MTOT * DIM) / 1024, 256, 0, stream>>>(x, xb, MTOT * DIM);

    gemm_bf16_kernel<false, false><<<dim3(MTOT / 128, DIM / 128), 256, 0, stream>>>(
        xb, Wqb, bq, (void*)Qb, MTOT);
    gemm_bf16_kernel<false, false><<<dim3(MTOT / 128, DIM / 128), 256, 0, stream>>>(
        xb, Wkb, bk, (void*)Kb, MTOT);
    gemm_bf16_kernel<true, false><<<dim3(MTOT / 128, DIM / 128), 256, 0, stream>>>(
        xb, Wvb, bv, (void*)Vt, MTOT);

    flash_attn_kernel<<<dim3(SEQ / 16, HEADS, BATCH), 32, 0, stream>>>(Qb, Kb, Vt, Yb);

    gemm_bf16_kernel<false, true><<<dim3(MTOT / 128, DIM / 128), 256, 0, stream>>>(
        Yb, Wpb, bp, (void*)out, MTOT);
}
